// ContextBlock_49658411876448
// MI455X (gfx1250) — matrix operand order repaired and checked by reading, not compiled, not run on hardware
//
#include <hip/hip_runtime.h>

// ---------------- problem constants ----------------
#define BB   8
#define LL   4
#define CLC  256
#define HWN  4096
#define CC   1024   // LL*CLC
#define PP   256
#define EPSF 1e-5f

typedef __attribute__((ext_vector_type(16))) _Float16 v16h;
typedef __attribute__((ext_vector_type(8)))  _Float16 v8h;
typedef __attribute__((ext_vector_type(8)))  float    v8f;

// ---------------- WMMA helpers ----------------
__device__ __forceinline__ v8f wmma_acc(v16h a, v16h b, v8f c) {
    // D = A(16x32 f16) x B(32x16 f16) + C(16x16 f32)
    return __builtin_amdgcn_wmma_f32_16x16x32_f16(
        /*neg_a=*/false, a, /*neg_b=*/false, b,
        /*c_mod=*/(short)0, c, /*reuse_a=*/false, /*reuse_b=*/false);
}

// A-matrix fragment (16x32 f16) from LDS, row-major [16, stride].
// Layout: lane m = lane&15; elems 0..15 -> K = kbase+off+0..15,
// off = (lane<16 ? 0 : 16).
__device__ __forceinline__ v16h load_afrag(const _Float16* base, int row,
                                           int stride, int k0, int lane) {
    int koff = (lane < 16) ? 0 : 16;
    const v8h* lo = reinterpret_cast<const v8h*>(base + row * stride + k0 + koff);
    const v8h* hi = reinterpret_cast<const v8h*>(base + row * stride + k0 + 8 + koff);
    v8h a = *lo, b = *hi;
    return __builtin_shufflevector(a, b, 0, 1, 2, 3, 4, 5, 6, 7,
                                   8, 9, 10, 11, 12, 13, 14, 15);
}

// B-matrix fragment (32x16 f16) where B[k][n] = W[n][k], W f32 row-major
// [*, stride]. Layout: lane n = lane&15; elems e -> K = kb + e,
// kb = (lane<16 ? 0 : 16). Per lane: 16 contiguous f32 -> 4x float4 loads.
__device__ __forceinline__ v16h load_bfrag_f32(const float* __restrict__ W,
                                               int nbase, int stride, int k0,
                                               int lane) {
    int n  = lane & 15;
    int kb = (lane < 16) ? 0 : 16;
    const float4* p4 = reinterpret_cast<const float4*>(
        W + (size_t)(nbase + n) * stride + k0 + kb);
    float4 w0 = p4[0], w1 = p4[1], w2 = p4[2], w3 = p4[3];
    v16h r;
    r[0]  = (_Float16)w0.x; r[1]  = (_Float16)w0.y;
    r[2]  = (_Float16)w0.z; r[3]  = (_Float16)w0.w;
    r[4]  = (_Float16)w1.x; r[5]  = (_Float16)w1.y;
    r[6]  = (_Float16)w1.z; r[7]  = (_Float16)w1.w;
    r[8]  = (_Float16)w2.x; r[9]  = (_Float16)w2.y;
    r[10] = (_Float16)w2.z; r[11] = (_Float16)w2.w;
    r[12] = (_Float16)w3.x; r[13] = (_Float16)w3.y;
    r[14] = (_Float16)w3.z; r[15] = (_Float16)w3.w;
    return r;
}

// ---------------- K1: partial logits (C split in 8 chunks of 128) ----------
// part[(chunk*B + b)*HW + n] = sum_{j<128} wm[chunk*128+j] * x[b, c, n]
__global__ __launch_bounds__(256) void k_logits_part(
    const float* __restrict__ x0, const float* __restrict__ x1,
    const float* __restrict__ x2, const float* __restrict__ x3,
    const float* __restrict__ wm, float* __restrict__ part) {
    int t = blockIdx.x * 256 + threadIdx.x;   // [0, 8*8*4096)
    int n     = t & (HWN - 1);
    int b     = (t >> 12) & 7;
    int chunk = t >> 15;                       // [0,8)
    int lev    = chunk >> 1;
    int clbase = (chunk & 1) * 128;
    const float* xp = (lev == 0) ? x0 : (lev == 1) ? x1 : (lev == 2) ? x2 : x3;
    const float* px = xp + ((size_t)(b * CLC + clbase)) * HWN + n;
    const float* pw = wm + lev * CLC + clbase;
    float acc = 0.f;
#pragma unroll 8
    for (int j = 0; j < 128; ++j)
        acc = fmaf(pw[j], px[(size_t)j * HWN], acc);
    part[t] = acc;
}

// ---------------- K2: sum chunks + softmax -> attn -----------------------
__global__ __launch_bounds__(256) void k_softmax(
    const float* __restrict__ part, const float* __restrict__ bm,
    float* __restrict__ attn) {
    __shared__ float red[256];
    int b = blockIdx.x, tid = threadIdx.x;
    float vals[16];
    float bias = bm[0];
    float lmax = -3.4e38f;
#pragma unroll
    for (int j = 0; j < 16; ++j) {
        int n = tid + j * 256;
        float s = bias;
#pragma unroll
        for (int ch = 0; ch < 8; ++ch)
            s += part[((size_t)(ch * BB + b) << 12) + n];
        vals[j] = s;
        lmax = fmaxf(lmax, s);
    }
    red[tid] = lmax;
    __syncthreads();
    for (int sft = 128; sft > 0; sft >>= 1) {
        if (tid < sft) red[tid] = fmaxf(red[tid], red[tid + sft]);
        __syncthreads();
    }
    float gmax = red[0];
    __syncthreads();
    float lsum = 0.f;
#pragma unroll
    for (int j = 0; j < 16; ++j) {
        vals[j] = __expf(vals[j] - gmax);
        lsum += vals[j];
    }
    red[tid] = lsum;
    __syncthreads();
    for (int sft = 128; sft > 0; sft >>= 1) {
        if (tid < sft) red[tid] += red[tid + sft];
        __syncthreads();
    }
    float inv = 1.f / red[0];
#pragma unroll
    for (int j = 0; j < 16; ++j)
        attn[(size_t)b * HWN + tid + j * 256] = vals[j] * inv;
}

// ---------------- K3: context[b,c] = sum_n x[b,c,n]*attn[b,n] -------------
__global__ __launch_bounds__(256) void k_context(
    const float* __restrict__ x0, const float* __restrict__ x1,
    const float* __restrict__ x2, const float* __restrict__ x3,
    const float* __restrict__ attn, float* __restrict__ ctx) {
    int wave = threadIdx.x >> 5, lane = threadIdx.x & 31;
    int g = blockIdx.x * 8 + wave;          // [0, B*C)
    int b = g >> 10, c = g & (CC - 1);
    int lev = c >> 8, cl = c & 255;
    const float* xp = (lev == 0) ? x0 : (lev == 1) ? x1 : (lev == 2) ? x2 : x3;
    const float4* p = reinterpret_cast<const float4*>(
        xp + ((size_t)(b * CLC + cl)) * HWN);
    const float4* a = reinterpret_cast<const float4*>(attn + (size_t)b * HWN);
    float sum = 0.f;
#pragma unroll 4
    for (int i = lane; i < HWN / 4; i += 32) {
        float4 xv = p[i], av = a[i];
        sum += xv.x * av.x + xv.y * av.y + xv.z * av.z + xv.w * av.w;
    }
#pragma unroll
    for (int m = 16; m >= 1; m >>= 1) sum += __shfl_xor(sum, m, 32);
    if (lane == 0) ctx[g] = sum;
}

// ---------------- K4/K5: gate MLP via WMMA --------------------------------
// One block per (branch, level): 8 blocks x 512 threads (16 waves).
// GEMM1: h[16,256] = vpad[16,1024] x W1^T, +b1, LayerNorm(p), ReLU
// GEMM2: o[16,256] = hn[16,256] x W2^T, +b2, (sigmoid), (+prev)
__global__ __launch_bounds__(512) void k_gate(
    const float* __restrict__ vin_add, const float* __restrict__ vin_mul,
    const float* __restrict__ aW1, const float* __restrict__ ab1,
    const float* __restrict__ ag,  const float* __restrict__ abe,
    const float* __restrict__ aW2, const float* __restrict__ ab2,
    const float* __restrict__ mW1, const float* __restrict__ mb1,
    const float* __restrict__ mg,  const float* __restrict__ mbe,
    const float* __restrict__ mW2, const float* __restrict__ mb2,
    const float* __restrict__ prev_add, const float* __restrict__ prev_mul,
    float* __restrict__ out_add, float* __restrict__ out_mul,
    int r, int use_prev) {
    __shared__ alignas(16) _Float16 vsh[16 * CC];   // 32 KB
    __shared__ alignas(16) _Float16 hsh[16 * PP];   //  8 KB
    __shared__ float psum[16][16], psq[16][16];
    __shared__ float redsum[16], redsq[16];

    const int branch = blockIdx.x >> 2;  // 0 = add, 1 = mul
    const int l      = blockIdx.x & 3;
    const int tid  = threadIdx.x;
    const int wave = tid >> 5, lane = tid & 31;

    const float* vin  = branch ? vin_mul : vin_add;
    const float* W1   = (branch ? mW1 : aW1) + (size_t)(r * LL + l) * PP * CC;
    const float* b1   = (branch ? mb1 : ab1) + (r * LL + l) * PP;
    const float* gg   = (branch ? mg  : ag ) + (r * LL + l) * PP;
    const float* be   = (branch ? mbe : abe) + (r * LL + l) * PP;
    const float* W2   = (branch ? mW2 : aW2) + (size_t)(r * LL + l) * CLC * PP;
    const float* b2   = (branch ? mb2 : ab2) + (r * LL + l) * CLC;
    const float* prev = branch ? prev_mul : prev_add;
    float*       outp = branch ? out_mul : out_add;

    // stage v [B,C] f32 -> vsh [16,C] f16, rows 8..15 zero-padded
    for (int i = tid; i < 16 * CC; i += 512) {
        int row = i >> 10, col = i & (CC - 1);
        vsh[i] = (row < BB) ? (_Float16)vin[row * CC + col] : (_Float16)0.f;
    }
    __syncthreads();

    // ---- GEMM1: 16 waves, wave owns p-tile [wave*16, wave*16+16) ----
    const int pbase = wave * 16;
    v8f acc = {};
    for (int k0 = 0; k0 < CC; k0 += 32) {
        v16h a  = load_afrag(vsh, lane & 15, CC, k0, lane);
        v16h bf = load_bfrag_f32(W1, pbase, CC, k0, lane);
        acc = wmma_acc(a, bf, acc);
    }
    float bias1 = b1[pbase + (lane & 15)];
    float h[8], s[8], q[8];
#pragma unroll
    for (int v = 0; v < 8; ++v) {
        h[v] = acc[v] + bias1;
        s[v] = h[v];
        q[v] = h[v] * h[v];
    }
    // deterministic LN reduction: reduce 16-lane groups, stash per-wave partials
#pragma unroll
    for (int m = 1; m < 16; m <<= 1) {
#pragma unroll
        for (int v = 0; v < 8; ++v) {
            s[v] += __shfl_xor(s[v], m, 32);
            q[v] += __shfl_xor(q[v], m, 32);
        }
    }
    if ((lane & 15) == 0) {
#pragma unroll
        for (int v = 0; v < 8; ++v) {
            int row = v + ((lane < 16) ? 0 : 8);
            psum[wave][row] = s[v];
            psq[wave][row]  = q[v];
        }
    }
    __syncthreads();
    if (tid < 16) {
        float ss = 0.f, qq = 0.f;
        for (int w = 0; w < 16; ++w) { ss += psum[w][tid]; qq += psq[w][tid]; }
        redsum[tid] = ss;
        redsq[tid]  = qq;
    }
    __syncthreads();
    float gv = gg[pbase + (lane & 15)], bev = be[pbase + (lane & 15)];
#pragma unroll
    for (int v = 0; v < 8; ++v) {
        int row  = v + ((lane < 16) ? 0 : 8);
        float mu  = redsum[row] * (1.f / PP);
        float var = redsq[row] * (1.f / PP) - mu * mu;
        float x = (h[v] - mu) * rsqrtf(var + EPSF) * gv + bev;
        x = (x > 0.f) ? x : 0.f;  // ReLU
        hsh[row * PP + pbase + (lane & 15)] = (_Float16)x;
    }
    __syncthreads();

    // ---- GEMM2: wave owns o-tile [wave*16, wave*16+16) ----
    const int obase = wave * 16;
    v8f acc2 = {};
    for (int k0 = 0; k0 < PP; k0 += 32) {
        v16h a  = load_afrag(hsh, lane & 15, PP, k0, lane);
        v16h bf = load_bfrag_f32(W2, obase, PP, k0, lane);
        acc2 = wmma_acc(a, bf, acc2);
    }
    if (lane < 16) {  // rows m = v (0..7) valid; lanes>=16 hold pad rows
        int o = obase + lane;
        float bias2 = b2[o];
#pragma unroll
        for (int v = 0; v < 8; ++v) {
            float val = acc2[v] + bias2;
            if (branch) val = 1.f / (1.f + __expf(-val));  // sigmoid (mul)
            int idx = v * CC + l * CLC + o;
            outp[idx] = use_prev ? (prev[idx] + val) : val;
        }
    }
}

// ---------------- K6: out = sum_l x_l*mul[b,l,cl] + sum_l add[b,l,cl] -----
__global__ __launch_bounds__(256) void k_apply(
    const float* __restrict__ x0, const float* __restrict__ x1,
    const float* __restrict__ x2, const float* __restrict__ x3,
    const float* __restrict__ mulF, const float* __restrict__ addF,
    float* __restrict__ out) {
    int b = blockIdx.x >> 8, cl = blockIdx.x & 255;
    size_t base = (size_t)(b * CLC + cl) * (HWN / 4);
    const float4* p0 = reinterpret_cast<const float4*>(x0) + base;
    const float4* p1 = reinterpret_cast<const float4*>(x1) + base;
    const float4* p2 = reinterpret_cast<const float4*>(x2) + base;
    const float4* p3 = reinterpret_cast<const float4*>(x3) + base;
    float4* po = reinterpret_cast<float4*>(out) + base;
    size_t mb = (size_t)b * CC + cl;
    float m0 = mulF[mb], m1 = mulF[mb + CLC], m2 = mulF[mb + 2 * CLC],
          m3 = mulF[mb + 3 * CLC];
    float aa = addF[mb] + addF[mb + CLC] + addF[mb + 2 * CLC] +
               addF[mb + 3 * CLC];
#pragma unroll 2
    for (int t = threadIdx.x; t < HWN / 4; t += 256) {
        float4 v0 = p0[t], v1 = p1[t], v2 = p2[t], v3 = p3[t];
        float4 o;
        o.x = v0.x * m0 + v1.x * m1 + v2.x * m2 + v3.x * m3 + aa;
        o.y = v0.y * m0 + v1.y * m1 + v2.y * m2 + v3.y * m3 + aa;
        o.z = v0.z * m0 + v1.z * m1 + v2.z * m2 + v3.z * m3 + aa;
        o.w = v0.w * m0 + v1.w * m1 + v2.w * m2 + v3.w * m3 + aa;
        po[t] = o;
    }
}

// ---------------- launcher ----------------
extern "C" void kernel_launch(void* const* d_in, const int* in_sizes, int n_in,
                              void* d_out, int out_size, void* d_ws,
                              size_t ws_size, hipStream_t stream) {
    (void)in_sizes; (void)n_in; (void)out_size; (void)ws_size;
    const float* x0  = (const float*)d_in[0];
    const float* x1  = (const float*)d_in[1];
    const float* x2  = (const float*)d_in[2];
    const float* x3  = (const float*)d_in[3];
    const float* wm  = (const float*)d_in[4];
    const float* bm  = (const float*)d_in[5];
    const float* aW1 = (const float*)d_in[6];
    const float* ab1 = (const float*)d_in[7];
    const float* ag  = (const float*)d_in[8];
    const float* abe = (const float*)d_in[9];
    const float* aW2 = (const float*)d_in[10];
    const float* ab2 = (const float*)d_in[11];
    const float* mW1 = (const float*)d_in[12];
    const float* mb1 = (const float*)d_in[13];
    const float* mg  = (const float*)d_in[14];
    const float* mbe = (const float*)d_in[15];
    const float* mW2 = (const float*)d_in[16];
    const float* mb2 = (const float*)d_in[17];

    float* ws    = (float*)d_ws;
    float* part  = ws;                       //  8*B*HW  = 262144
    float* attn  = part + 8 * BB * HWN;      //  B*HW    =  32768
    float* ctx   = attn + BB * HWN;          //  B*C     =   8192
    float* addM0 = ctx + BB * CC;            //  8192
    float* mulM0 = addM0 + BB * CC;          //  8192
    float* addF  = mulM0 + BB * CC;          //  8192
    float* mulF  = addF + BB * CC;           //  8192
    float* out   = (float*)d_out;

    k_logits_part<<<(8 * BB * HWN) / 256, 256, 0, stream>>>(x0, x1, x2, x3, wm,
                                                            part);
    k_softmax<<<BB, 256, 0, stream>>>(part, bm, attn);
    k_context<<<(BB * CC) / 8, 256, 0, stream>>>(x0, x1, x2, x3, attn, ctx);
    k_gate<<<8, 512, 0, stream>>>(ctx, ctx, aW1, ab1, ag, abe, aW2, ab2, mW1,
                                  mb1, mg, mbe, mW2, mb2, nullptr, nullptr,
                                  addM0, mulM0, /*r=*/0, /*use_prev=*/0);
    k_gate<<<8, 512, 0, stream>>>(addM0, mulM0, aW1, ab1, ag, abe, aW2, ab2,
                                  mW1, mb1, mg, mbe, mW2, mb2, addM0, mulM0,
                                  addF, mulF, /*r=*/1, /*use_prev=*/1);
    k_apply<<<BB * CLC, 256, 0, stream>>>(x0, x1, x2, x3, mulF, addF, out);
}
